// FullyConnectedLoCS_79491254714575
// MI455X (gfx1250) — matrix operand order repaired and checked by reading, not compiled, not run on hardware
//
#include <hip/hip_runtime.h>
#include <math.h>

#define B_  64
#define N_  100
#define H_  128
#define EPN 99      // senders per receiver (N-1)
#define MPAD 112    // 99 edges padded to 7 tiles of 16
#define K1  32      // edge-attr K (11) padded to one WMMA K-step

typedef __attribute__((ext_vector_type(16))) _Float16 v16h;
typedef __attribute__((ext_vector_type(8)))  float    v8f;

// native-precision silu: x * sigmoid(x), via v_exp_f32 (2^x) + v_rcp_f32
__device__ __forceinline__ float fast_silu(float x) {
  const float e = __builtin_amdgcn_exp2f(-x * 1.4426950408889634f);
  return x * __builtin_amdgcn_rcpf(1.f + e);
}

// ---- WMMA fragment access ----
// A 16x32 f16 tile (this code's contiguous K order, matched by B; not the ISA table order), from row-major [rows][ld] f16 LDS:
// lane = row (mod 16); half h -> K = (lane<16?0:16)+h.
__device__ __forceinline__ v16h frag_a(const _Float16* s, int ld, int mt, int kt, int lane) {
  const _Float16* row = s + (mt * 16 + (lane & 15)) * ld + kt * 32 + ((lane >> 4) << 4);
  v16h a;
#pragma unroll
  for (int h = 0; h < 8; ++h) a[h] = row[h];
#pragma unroll
  for (int h = 0; h < 8; ++h) a[8 + h] = row[8 + h];
  return a;
}

// B fragments are stored PRE-SWIZZLED in LDS: [tile(kt*8+nt)][lane][16 halves],
// so a fragment load is one 32-byte contiguous vector read (2x ds_load_b128).
// B 32x16 layout: lane = col (mod 16) | (K>=16)<<4 ; half h -> K = base + h.
__device__ __forceinline__ v16h frag_b_swz(const _Float16* s, int kt, int nt, int lane) {
  return *(const v16h*)(s + (((((kt << 3) | nt) << 5) | lane) << 4));
}

// convert+swizzle a 128x128 f32 weight matrix (row-major [K][N]) into LDS
__device__ __forceinline__ void load_w_swz(const float* __restrict__ w, _Float16* dst, int tid) {
  for (int idx = tid; idx < H_ * H_; idx += 256) {
    const int k = idx >> 7, n = idx & (H_ - 1);
    const int tile = ((k >> 5) << 3) | (n >> 4);
    const int lane = (n & 15) | (((k >> 4) & 1) << 4);
    dst[(((tile << 5) | lane) << 4) | (k & 15)] = (_Float16)w[idx];
  }
}

#define WMMA_F16(A, Bf, C) \
  __builtin_amdgcn_wmma_f32_16x16x32_f16(false, (A), false, (Bf), (short)0, (C), false, false)

// ---------------------------------------------------------------------------
// Kernel 1: per (batch, receiver) workgroup.
// Builds 99 edge-attr rows (11 feats, f16, zero-padded to 112x32) in LDS,
// GEMM (112x32)@(32x128) with WMMA, silu, masked column-sum over edges,
// writes agg_pre[node][128] = sum_j silu(ea_ij @ W1 + b1).
// ---------------------------------------------------------------------------
__global__ void __launch_bounds__(256)
locs_edge_kernel(const float* __restrict__ inp, const float* __restrict__ ef_w1,
                 const float* __restrict__ ef_b1, float* __restrict__ aggpre) {
  __shared__ __align__(64) _Float16 eah[MPAD * K1];  // 7 KiB, row-major A
  __shared__ __align__(64) _Float16 w1h[8 * 32 * 16];// 8 KiB, swizzled B tiles
  const int tid  = threadIdx.x;
  const int lane = tid & 31;
  const int wave = tid >> 5;
  const int node = blockIdx.x;          // b * N + i
  const int b = node / N_;
  const int i = node - b * N_;

  for (int idx = tid; idx < MPAD * K1; idx += 256) eah[idx] = (_Float16)0.f;
  // ef_w1 is 11x128; zero-pad K to 32 and store swizzled (kt==0 only)
  for (int idx = tid; idx < K1 * H_; idx += 256) {
    const int k = idx >> 7, n = idx & (H_ - 1);
    const float v = (k < 11) ? ef_w1[k * H_ + n] : 0.f;
    const int lane_d = (n & 15) | (((k >> 4) & 1) << 4);
    w1h[(((((n >> 4) << 5) | lane_d)) << 4) | (k & 15)] = (_Float16)v;
  }

  // receiver local frame (cos/sin of velocity angle; atan2(0,0)=0 -> c=1,s=0)
  const float* xi = inp + (size_t)node * 4;
  const float pix = xi[0], piy = xi[1], vix = xi[2], viy = xi[3];
  const float ri = sqrtf(vix * vix + viy * viy);
  const float ci = (ri > 0.f) ? vix / ri : 1.f;
  const float si = (ri > 0.f) ? viy / ri : 0.f;

  __syncthreads();

  if (tid < EPN) {
    const int j = tid + (tid >= i);     // sender, skipping the diagonal
    const float* xj = inp + ((size_t)b * N_ + j) * 4;
    const float pjx = xj[0], pjy = xj[1], vjx = xj[2], vjy = xj[3];
    const float rj = sqrtf(vjx * vjx + vjy * vjy);
    const float cj = (rj > 0.f) ? vjx / rj : 1.f;
    const float sj = (rj > 0.f) ? vjy / rj : 0.f;
    const float dx = pjx - pix, dy = pjy - piy;
    const float rrp0 =  ci * dx + si * dy;        // R_i^T (p_j - p_i)
    const float rrp1 = -si * dx + ci * dy;
    const float sd = sj * ci - cj * si;           // sin(th_j - th_i)
    const float cd = cj * ci + sj * si;           // cos(th_j - th_i)
    const float euler = atan2f(sd, cd) * 0.3183098861837907f;
    const float dist  = sqrtf(dx * dx + dy * dy);
    const float theta = atan2f(rrp1, rrp0);
    const float rv0 =  ci * vjx + si * vjy;       // R_i^T v_j
    const float rv1 = -si * vjx + ci * vjy;
    _Float16* row = eah + tid * K1;
    row[0] = (_Float16)rrp0; row[1] = (_Float16)rrp1; row[2] = (_Float16)euler;
    row[3] = (_Float16)dist; row[4] = (_Float16)theta;
    row[5] = (_Float16)rv0;  row[6] = (_Float16)rv1;
    row[9] = (_Float16)ri;   // rel_feat[recv] = [0,0,|v_i|,0]; 7,8,10 stay 0
  }
  __syncthreads();

  // wave w owns output columns [16w, 16w+16)
  const int n = wave * 16 + (lane & 15);
  const float bias = ef_b1[n];
  const v16h bf = frag_b_swz(w1h, 0, wave, lane);
  float colsum = 0.f;
#pragma unroll
  for (int mt = 0; mt < 7; ++mt) {
    v8f acc = {};
    acc = WMMA_F16(frag_a(eah, K1, mt, 0, lane), bf, acc);
#pragma unroll
    for (int r = 0; r < 8; ++r) {
      const int m = mt * 16 + r + ((lane >> 4) << 3);
      if (m < EPN)                      // mask zero-padded rows (silu(b1)!=0)
        colsum += fast_silu(acc[r] + bias);
    }
  }
  colsum += __shfl_xor(colsum, 16, 32); // combine the two M-halves (wave32)
  if (lane < 16) aggpre[(size_t)node * H_ + n] = colsum;
}

// shared GEMM+ReLU stage for the node MLP: Y = relu(X @ W + bias), 64x128
__device__ __forceinline__ void gemm_relu(const _Float16* X, const _Float16* W,
                                          _Float16* Y, const float* __restrict__ bias,
                                          int wave, int lane) {
  const int n = wave * 16 + (lane & 15);
  const float bv = bias[n];
  const v16h b0 = frag_b_swz(W, 0, wave, lane);
  const v16h b1 = frag_b_swz(W, 1, wave, lane);
  const v16h b2 = frag_b_swz(W, 2, wave, lane);
  const v16h b3 = frag_b_swz(W, 3, wave, lane);
#pragma unroll
  for (int mt = 0; mt < 4; ++mt) {
    v8f acc = {};
    acc = WMMA_F16(frag_a(X, H_, mt, 0, lane), b0, acc);
    acc = WMMA_F16(frag_a(X, H_, mt, 1, lane), b1, acc);
    acc = WMMA_F16(frag_a(X, H_, mt, 2, lane), b2, acc);
    acc = WMMA_F16(frag_a(X, H_, mt, 3, lane), b3, acc);
#pragma unroll
    for (int r = 0; r < 8; ++r) {
      const int m = mt * 16 + r + ((lane >> 4) << 3);
      const float v = acc[r] + bv;
      Y[m * H_ + n] = (_Float16)((v > 0.f) ? v : 0.f);
    }
  }
}

// ---------------------------------------------------------------------------
// Kernel 2: node MLP chain over 64-node row blocks.
// aug = (agg_pre/99) @ ef_w2 + ef_b2 + |v|*res_w[2] + res_b
// y = relu(aug@out_w1+b); y = relu(y@out_w2+b); pred = y@out_w3+b
// out = inputs + [R pred[0:2], R pred[2:4]]
// ---------------------------------------------------------------------------
__global__ void __launch_bounds__(256)
locs_node_kernel(const float* __restrict__ inp, const float* __restrict__ aggpre,
                 const float* __restrict__ ef_w2, const float* __restrict__ ef_b2,
                 const float* __restrict__ res_w, const float* __restrict__ res_b,
                 const float* __restrict__ out_w1, const float* __restrict__ out_b1,
                 const float* __restrict__ out_w2, const float* __restrict__ out_b2,
                 const float* __restrict__ out_w3, const float* __restrict__ out_b3,
                 float* __restrict__ out) {
  __shared__ __align__(64) unsigned char smem[65536];
  _Float16* Xs = (_Float16*)smem;             // 64x128 f16 row-major (16 KiB)
  _Float16* Ys = (_Float16*)(smem + 16384);   // 64x128 f16 row-major (16 KiB)
  _Float16* Ws = (_Float16*)(smem + 32768);   // swizzled B tiles (32 KiB)
  float*  predS = (float*)(smem + 32768);     // reused after last GEMM (1 KiB)

  const int tid  = threadIdx.x;
  const int lane = tid & 31;
  const int wave = tid >> 5;
  const int base = blockIdx.x * 64;
  const float inv99 = 1.f / 99.f;

  for (int idx = tid; idx < 64 * H_; idx += 256)
    Xs[idx] = (_Float16)(aggpre[(size_t)base * H_ + idx] * inv99);
  load_w_swz(ef_w2, Ws, tid);
  __syncthreads();

  // ---- stage A: aug (custom epilogue, no activation) -> Ys ----
  {
    const int n = wave * 16 + (lane & 15);
    const float badd = ef_b2[n] + res_b[n];
    const float rw = res_w[2 * H_ + n];       // rel_feat = [0,0,|v|,0]
    const v16h b0 = frag_b_swz(Ws, 0, wave, lane);
    const v16h b1 = frag_b_swz(Ws, 1, wave, lane);
    const v16h b2 = frag_b_swz(Ws, 2, wave, lane);
    const v16h b3 = frag_b_swz(Ws, 3, wave, lane);
#pragma unroll
    for (int mt = 0; mt < 4; ++mt) {
      v8f acc = {};
      acc = WMMA_F16(frag_a(Xs, H_, mt, 0, lane), b0, acc);
      acc = WMMA_F16(frag_a(Xs, H_, mt, 1, lane), b1, acc);
      acc = WMMA_F16(frag_a(Xs, H_, mt, 2, lane), b2, acc);
      acc = WMMA_F16(frag_a(Xs, H_, mt, 3, lane), b3, acc);
#pragma unroll
      for (int r = 0; r < 8; ++r) {
        const int m = mt * 16 + r + ((lane >> 4) << 3);
        const int nd = base + m;
        const float vx = inp[nd * 4 + 2], vy = inp[nd * 4 + 3];
        const float nv = sqrtf(vx * vx + vy * vy);
        Ys[m * H_ + n] = (_Float16)(acc[r] + badd + nv * rw);
      }
    }
  }
  __syncthreads();

  // ---- stage B: relu(Ys @ out_w1 + out_b1) -> Xs ----
  load_w_swz(out_w1, Ws, tid);
  __syncthreads();
  gemm_relu(Ys, Ws, Xs, out_b1, wave, lane);
  __syncthreads();

  // ---- stage C: relu(Xs @ out_w2 + out_b2) -> Ys ----
  load_w_swz(out_w2, Ws, tid);
  __syncthreads();
  gemm_relu(Xs, Ws, Ys, out_b2, wave, lane);
  __syncthreads();

  // ---- stage D: 128->4 head (VALU), rotate back, add residual ----
  {
    const int m = tid >> 2, o = tid & 3;      // 64 rows x 4 outputs
    float acc = out_b3[o];
    const _Float16* yr = Ys + m * H_;
#pragma unroll 4
    for (int k = 0; k < H_; ++k) acc += (float)yr[k] * out_w3[k * 4 + o];
    predS[tid] = acc;                         // predS[m*4+o]; Ws region is free
  }
  __syncthreads();
  {
    const int m = tid >> 2, o = tid & 3;
    const int nd = base + m;
    const float vx = inp[nd * 4 + 2], vy = inp[nd * 4 + 3];
    const float r = sqrtf(vx * vx + vy * vy);
    const float c = (r > 0.f) ? vx / r : 1.f;
    const float s = (r > 0.f) ? vy / r : 0.f;
    const int pb = m * 4 + (o & 2);           // pair base (pos pair / vel pair)
    const float p0 = predS[pb], p1 = predS[pb + 1];
    const float rot = (o & 1) ? (s * p0 + c * p1) : (c * p0 - s * p1);
    out[nd * 4 + o] = inp[nd * 4 + o] + rot;
  }
}

extern "C" void kernel_launch(void* const* d_in, const int* in_sizes, int n_in,
                              void* d_out, int out_size, void* d_ws, size_t ws_size,
                              hipStream_t stream) {
  const float* inp    = (const float*)d_in[0];
  const float* ef_w1  = (const float*)d_in[1];
  const float* ef_b1  = (const float*)d_in[2];
  const float* ef_w2  = (const float*)d_in[3];
  const float* ef_b2  = (const float*)d_in[4];
  const float* res_w  = (const float*)d_in[5];
  const float* res_b  = (const float*)d_in[6];
  const float* out_w1 = (const float*)d_in[7];
  const float* out_b1 = (const float*)d_in[8];
  const float* out_w2 = (const float*)d_in[9];
  const float* out_b2 = (const float*)d_in[10];
  const float* out_w3 = (const float*)d_in[11];
  const float* out_b3 = (const float*)d_in[12];

  float* aggpre = (float*)d_ws;   // B*N*H f32 = 3.125 MiB of scratch

  locs_edge_kernel<<<B_ * N_, 256, 0, stream>>>(inp, ef_w1, ef_b1, aggpre);
  locs_node_kernel<<<(B_ * N_) / 64, 256, 0, stream>>>(
      inp, aggpre, ef_w2, ef_b2, res_w, res_b,
      out_w1, out_b1, out_w2, out_b2, out_w3, out_b3, (float*)d_out);
}
